// AdaptiveMultiLoRALinear_65970697666720
// MI455X (gfx1250) — compile-verified, matrix operand order checked by reading, not run on hardware
//
#include <hip/hip_runtime.h>

// ---------------------------------------------------------------------------
// AdaptiveMultiLoRALinear for MI455X (gfx1250, wave32, WMMA + TDM)
//
// z = x @ W^T + b                      [8192, 4096] fp32
// m = sum_e p_e * (x @ A_e @ B_e)      via U = x@A (K=4096), m = U@(pB)^T (K=128)
// out = z + min(0.5*|z|/(|m|+eps), 1) * m   (row norms over O)
//
// Precision: split-bf16 (hi/lo) fp32 emulation, 3x v_wmma_f32_16x16x32_bf16.
// Data movement (gemm_z): Tensor Data Mover double-buffered LDS staging,
// synchronized with s_wait_tensorcnt + workgroup barriers.
// ---------------------------------------------------------------------------

#define DEV __device__ __forceinline__

typedef __bf16 v16bf __attribute__((ext_vector_type(16)));
typedef __bf16 v8bf  __attribute__((ext_vector_type(8)));
typedef float  v8f   __attribute__((ext_vector_type(8)));
typedef unsigned int u32x4 __attribute__((ext_vector_type(4)));
typedef int          i32x4 __attribute__((ext_vector_type(4)));
typedef int          i32x8 __attribute__((ext_vector_type(8)));

#if __has_include(<hip/amd_detail/amd_gfx1250_TDM.h>)
#define TDM_SIX_ARGS 1   // amdgpu-toolchain (therock-10.0 headers): 6-arg builtin
#else
#define TDM_SIX_ARGS 0   // ROCm 7.2 clang-22: 5-arg builtin
#endif

static constexpr int Bb = 4, S = 2048, D = 4096, O = 4096, R = 16, E = 8;
static constexpr int NROWS = Bb * S;      // 8192
static constexpr int RC    = E * R;       // 128 (lora inner dim)
static constexpr float C_CLAMP = 0.5f;
static constexpr float EPSV    = 1e-6f;

union Frag16 { v16bf v; v8bf h[2]; };

// Per-lane WMMA fragment load from k-major storage. Lane L covers row/col
// (L&15); k-chunks {kb..kb+7} and {kb+16..kb+23}, kb = 8*(L>>4).
DEV v16bf load_frag(const __bf16* p) {
    Frag16 f;
    f.h[0] = *reinterpret_cast<const v8bf*>(p);
    f.h[1] = *reinterpret_cast<const v8bf*>(p + 16);
    return f.v;
}

DEV v8f wmma_bf16(v16bf a, v16bf b, v8f c) {
    return __builtin_amdgcn_wmma_f32_16x16x32_bf16(
        false, a, false, b, (short)0, c, false, false);
}

// ---------------------------------------------------------------------------
// TDM: 2D tile load (128 rows x 32 bf16) global -> LDS, with LDS padding of
// 4 DWORDs every 16 DWORDs => 80B LDS row stride (16B aligned frag chunks).
// Issued once per wave (EXEC-independent); tracked by TENSORcnt.
// ---------------------------------------------------------------------------
DEV void tdm_load_tile(unsigned int lds_addr, const __bf16* gtile,
                       unsigned int tensor_rows) {
    unsigned long long ga = (unsigned long long)(uintptr_t)gtile;
    u32x4 g0 = {
        1u,                                   // count=1 (valid user descriptor)
        lds_addr,                             // LDS byte address
        (unsigned int)ga,                     // global_addr[31:0]
        (unsigned int)((ga >> 32) & 0x01FFFFFFu) | 0x80000000u  // [56:32] | type=2
    };
    i32x8 g1 = {
        (int)((1u << 16) |                    // data_size = 2 bytes
              (1u << 20) |                    // pad_enable
              (3u << 22) |                    // pad_interval: 16 DWORDs (64B)
              (3u << 25)),                    // pad_amount: 4 DWORDs (16B)
        (int)(((unsigned)D & 0xFFFFu) << 16),               // tensor_dim0[15:0]
        (int)((tensor_rows & 0xFFFFu) << 16),               // tensor_dim1[15:0]
        (int)((32u << 16) | (tensor_rows >> 16)),           // tile_dim0=32 | dim1 hi
        128,                                                // tile_dim1 = 128 rows
        (int)D,                                             // tensor_dim0_stride
        0, 0
    };
    i32x4 z4 = {0, 0, 0, 0};
#if TDM_SIX_ARGS
    i32x8 z8 = {0, 0, 0, 0, 0, 0, 0, 0};
    __builtin_amdgcn_tensor_load_to_lds(g0, g1, z4, z4, z8, 0);
#else
    __builtin_amdgcn_tensor_load_to_lds(g0, g1, z4, z4, 0);
#endif
}

// ---------------------------------------------------------------------------
// Prep kernels: fp32 -> (hi, lo) bf16 splits, with transposes for B-operands
// ---------------------------------------------------------------------------

__global__ void split_bf16_kernel(const float* __restrict__ src,
                                  __bf16* __restrict__ hi,
                                  __bf16* __restrict__ lo, int n) {
    int i = blockIdx.x * blockDim.x + threadIdx.x;
    int stride = gridDim.x * blockDim.x;
    for (; i < n; i += stride) {
        float v = src[i];
        __bf16 h = (__bf16)v;
        hi[i] = h;
        lo[i] = (__bf16)(v - (float)h);
    }
}

// A[e][k][r] (fp32) -> At[(e*16+r)][k] (bf16 hi/lo), k-major for B-frags
__global__ void prep_At_kernel(const float* __restrict__ A,
                               __bf16* __restrict__ hi,
                               __bf16* __restrict__ lo) {
    int i = blockIdx.x * blockDim.x + threadIdx.x;   // i indexes (e,k,r)
    int e = i / (D * R);
    int rem = i - e * (D * R);
    int k = rem / R;
    int r = rem - k * R;
    float v = A[i];
    __bf16 h = (__bf16)v;
    int o = (e * R + r) * D + k;
    hi[o] = h;
    lo[o] = (__bf16)(v - (float)h);
}

// B[e][r][o] (fp32) scaled by p[e] -> Bst[o][e*16+r] (bf16 hi/lo), k-major
__global__ void prep_Bst_kernel(const float* __restrict__ Bm,
                                const float* __restrict__ p,
                                __bf16* __restrict__ hi,
                                __bf16* __restrict__ lo) {
    int i = blockIdx.x * blockDim.x + threadIdx.x;   // i indexes (e,r,o)
    int er = i / O;
    int o  = i - er * O;
    int e  = er / R;
    float v = p[e] * Bm[i];
    __bf16 h = (__bf16)v;
    int idx = o * RC + er;
    hi[idx] = h;
    lo[idx] = (__bf16)(v - (float)h);
}

// ---------------------------------------------------------------------------
// U = x @ A_e : per expert [8192x4096]@[4096x16], K-loop over 4096.
// One wave per 16-row tile; 8 expert accumulators; x frag shared across experts.
// ---------------------------------------------------------------------------

__global__ __launch_bounds__(128) void lora_u_kernel(
    const __bf16* __restrict__ xh, const __bf16* __restrict__ xl,
    const __bf16* __restrict__ Ath, const __bf16* __restrict__ Atl,
    __bf16* __restrict__ Uh, __bf16* __restrict__ Ul) {
    const int lane  = threadIdx.x & 31;
    const int w     = threadIdx.x >> 5;
    const int half  = lane >> 4;
    const int m     = lane & 15;
    const int kb    = half * 8;
    const int row0  = blockIdx.x * 64 + w * 16;

    v8f acc[E] = {};

    const __bf16* xrh = xh + (size_t)(row0 + m) * D + kb;
    const __bf16* xrl = xl + (size_t)(row0 + m) * D + kb;

    for (int kk = 0; kk < D; kk += 32) {
        v16bf ah = load_frag(xrh + kk);
        v16bf al = load_frag(xrl + kk);
        #pragma unroll
        for (int e = 0; e < E; ++e) {
            const size_t boff = (size_t)(e * 16 + m) * D + kk + kb;
            v16bf bh = load_frag(Ath + boff);
            v16bf bl = load_frag(Atl + boff);
            acc[e] = wmma_bf16(ah, bh, acc[e]);
            acc[e] = wmma_bf16(ah, bl, acc[e]);
            acc[e] = wmma_bf16(al, bh, acc[e]);
        }
    }

    #pragma unroll
    for (int e = 0; e < E; ++e) {
        #pragma unroll
        for (int v = 0; v < 8; ++v) {
            int row = row0 + v + 8 * half;   // C layout: VGPR v -> M=v (+8 hi half)
            int col = e * 16 + m;
            float val = acc[e][v];
            __bf16 h = (__bf16)val;
            Uh[(size_t)row * RC + col] = h;
            Ul[(size_t)row * RC + col] = (__bf16)(val - (float)h);
        }
    }
}

// ---------------------------------------------------------------------------
// z = x @ W^T + b. Workgroup = 8 waves, 128x128 tile; TDM double-buffered
// LDS staging (4 tiles/k-step: xh/xl/Wh/Wl, 128x32 bf16 each, 80B row pitch).
// Each wave computes 32x64 (2 M-tiles x 4 N-tiles), 24 WMMAs per k-step.
// Writes z to d_out and accumulates per-row sum(z^2).
// ---------------------------------------------------------------------------

static constexpr int LDSROW = 80;             // 64B data + 16B pad per row
static constexpr int TILEB  = 128 * LDSROW;   // 10240 B per tile

__global__ __launch_bounds__(256) void gemm_z_kernel(
    const __bf16* __restrict__ xh, const __bf16* __restrict__ xl,
    const __bf16* __restrict__ Wh, const __bf16* __restrict__ Wl,
    const float* __restrict__ bias,
    float* __restrict__ out, float* __restrict__ nz2) {
    __shared__ __align__(128) char smem[2 * 4 * TILEB];   // 80 KB, double buffer

    const int tid  = (int)threadIdx.x;
    const int w    = __builtin_amdgcn_readfirstlane(tid) >> 5;  // SGPR wave id
    const int lane = tid & 31;
    const int wr   = w & 3;
    const int wc   = w >> 2;
    const int half = lane >> 4;
    const int m    = lane & 15;
    const int kb   = half * 8;
    const int rowg = blockIdx.y * 128;            // workgroup row base
    const int colg = blockIdx.x * 128;            // workgroup col base
    const int row0 = rowg + wr * 32;
    const int col0 = colg + wc * 64;

    const unsigned int lds0 = (unsigned int)(uintptr_t)&smem[0];

    // One TDM descriptor per tile, issued by waves 0..3 (scalar-branched).
    auto stage = [&](int buf, int kk) {
        unsigned int base = lds0 + (unsigned)(buf * 4) * TILEB;
        if (w == 0)      tdm_load_tile(base + 0 * TILEB, xh + (size_t)rowg * D + kk, NROWS);
        else if (w == 1) tdm_load_tile(base + 1 * TILEB, xl + (size_t)rowg * D + kk, NROWS);
        else if (w == 2) tdm_load_tile(base + 2 * TILEB, Wh + (size_t)colg * D + kk, O);
        else if (w == 3) tdm_load_tile(base + 3 * TILEB, Wl + (size_t)colg * D + kk, O);
    };

    v8f acc[2][4] = {};

    stage(0, 0);
    int buf = 0;
    for (int kk = 0; kk < D; kk += 32, buf ^= 1) {
        if (kk + 32 < D) {
            stage(buf ^ 1, kk + 32);                      // prefetch next k-step
            if (w < 4) __builtin_amdgcn_s_wait_tensorcnt(1);  // current tile done
        } else {
            if (w < 4) __builtin_amdgcn_s_wait_tensorcnt(0);
        }
        __syncthreads();   // current buffer visible to all waves

        const char* cb = smem + buf * 4 * TILEB;
        const __bf16* xtH = (const __bf16*)(cb + 0 * TILEB);
        const __bf16* xtL = (const __bf16*)(cb + 1 * TILEB);
        const __bf16* wtH = (const __bf16*)(cb + 2 * TILEB);
        const __bf16* wtL = (const __bf16*)(cb + 3 * TILEB);
        constexpr int RE = LDSROW / 2;   // row pitch in elements

        v16bf ah[2], al[2], bh[4], bl[4];
        #pragma unroll
        for (int t = 0; t < 2; ++t) {
            const int r = (wr * 32 + t * 16 + m) * RE + kb;
            ah[t] = load_frag(xtH + r);
            al[t] = load_frag(xtL + r);
        }
        #pragma unroll
        for (int u = 0; u < 4; ++u) {
            const int r = (wc * 64 + u * 16 + m) * RE + kb;
            bh[u] = load_frag(wtH + r);
            bl[u] = load_frag(wtL + r);
        }
        #pragma unroll
        for (int t = 0; t < 2; ++t) {
            #pragma unroll
            for (int u = 0; u < 4; ++u) {
                acc[t][u] = wmma_bf16(ah[t], bh[u], acc[t][u]);
                acc[t][u] = wmma_bf16(ah[t], bl[u], acc[t][u]);
                acc[t][u] = wmma_bf16(al[t], bh[u], acc[t][u]);
            }
        }
        __syncthreads();   // frag reads done before next TDM overwrites buffer
    }

    #pragma unroll
    for (int t = 0; t < 2; ++t) {
        float s[8] = {};
        #pragma unroll
        for (int u = 0; u < 4; ++u) {
            int col = col0 + u * 16 + m;
            float bv = bias[col];
            #pragma unroll
            for (int v = 0; v < 8; ++v) {
                int row = row0 + t * 16 + v + 8 * half;
                float z = acc[t][u][v] + bv;
                out[(size_t)row * O + col] = z;
                s[v] += z * z;
            }
        }
        // reduce across the 16 lanes sharing each row (xor masks stay in-half)
        #pragma unroll
        for (int v = 0; v < 8; ++v) {
            float sv = s[v];
            sv += __shfl_xor(sv, 1);
            sv += __shfl_xor(sv, 2);
            sv += __shfl_xor(sv, 4);
            sv += __shfl_xor(sv, 8);
            if (m == 0) atomicAdd(&nz2[row0 + t * 16 + v + 8 * half], sv);
        }
    }
}

// ---------------------------------------------------------------------------
// m = U @ (p*B)^T, K=128 (operands fit in L2). phase 0: accumulate |m|^2.
// phase 1: recompute m and apply the gated blend in place.
// ---------------------------------------------------------------------------

__global__ __launch_bounds__(256) void lora_m_kernel(
    const __bf16* __restrict__ Uh, const __bf16* __restrict__ Ul,
    const __bf16* __restrict__ Bsh, const __bf16* __restrict__ Bsl,
    const float* __restrict__ nz2, float* __restrict__ nm2,
    float* __restrict__ out, int phase) {
    const int lane = threadIdx.x & 31;
    const int w    = threadIdx.x >> 5;
    const int half = lane >> 4;
    const int m    = lane & 15;
    const int kb   = half * 8;
    const int row0 = blockIdx.y * 128 + w * 16;
    const int col0 = blockIdx.x * 64;

    v8f acc[4] = {};

    #pragma unroll
    for (int kk = 0; kk < RC; kk += 32) {
        const size_t aoff = (size_t)(row0 + m) * RC + kk + kb;
        v16bf ah = load_frag(Uh + aoff);
        v16bf al = load_frag(Ul + aoff);
        #pragma unroll
        for (int u = 0; u < 4; ++u) {
            const size_t boff = (size_t)(col0 + u * 16 + m) * RC + kk + kb;
            v16bf bh = load_frag(Bsh + boff);
            v16bf bl = load_frag(Bsl + boff);
            acc[u] = wmma_bf16(ah, bh, acc[u]);
            acc[u] = wmma_bf16(ah, bl, acc[u]);
            acc[u] = wmma_bf16(al, bh, acc[u]);
        }
    }

    if (phase == 0) {
        float s[8] = {};
        #pragma unroll
        for (int u = 0; u < 4; ++u)
            #pragma unroll
            for (int v = 0; v < 8; ++v) s[v] += acc[u][v] * acc[u][v];
        #pragma unroll
        for (int v = 0; v < 8; ++v) {
            float sv = s[v];
            sv += __shfl_xor(sv, 1);
            sv += __shfl_xor(sv, 2);
            sv += __shfl_xor(sv, 4);
            sv += __shfl_xor(sv, 8);
            if (m == 0) atomicAdd(&nm2[row0 + v + 8 * half], sv);
        }
    } else {
        float g[8];
        #pragma unroll
        for (int v = 0; v < 8; ++v) {
            int row = row0 + v + 8 * half;
            float nz = sqrtf(nz2[row]);
            float nm = sqrtf(nm2[row]);
            g[v] = fminf(C_CLAMP * nz / (nm + EPSV), 1.0f);
        }
        #pragma unroll
        for (int u = 0; u < 4; ++u) {
            int col = col0 + u * 16 + m;
            #pragma unroll
            for (int v = 0; v < 8; ++v) {
                int row = row0 + v + 8 * half;
                size_t idx = (size_t)row * O + col;
                out[idx] = out[idx] + g[v] * acc[u][v];
            }
        }
    }
}

// ---------------------------------------------------------------------------

extern "C" void kernel_launch(void* const* d_in, const int* in_sizes, int n_in,
                              void* d_out, int out_size, void* d_ws, size_t ws_size,
                              hipStream_t stream) {
    const float* x    = (const float*)d_in[0];
    const float* W    = (const float*)d_in[1];
    const float* bias = (const float*)d_in[2];
    const float* A    = (const float*)d_in[3];
    const float* Bm   = (const float*)d_in[4];
    const float* p    = (const float*)d_in[5];
    float* out = (float*)d_out;

    char* ws = (char*)d_ws;
    size_t off = 0;
    auto alloc = [&](size_t bytes) -> void* {
        void* ptr = ws + off;
        off += (bytes + 255) & ~(size_t)255;
        return ptr;
    };

    __bf16* xh  = (__bf16*)alloc((size_t)NROWS * D * 2);
    __bf16* xl  = (__bf16*)alloc((size_t)NROWS * D * 2);
    __bf16* Wh  = (__bf16*)alloc((size_t)O * D * 2);
    __bf16* Wl  = (__bf16*)alloc((size_t)O * D * 2);
    __bf16* Ath = (__bf16*)alloc((size_t)E * R * D * 2);
    __bf16* Atl = (__bf16*)alloc((size_t)E * R * D * 2);
    __bf16* Bsh = (__bf16*)alloc((size_t)O * RC * 2);
    __bf16* Bsl = (__bf16*)alloc((size_t)O * RC * 2);
    __bf16* Uh  = (__bf16*)alloc((size_t)NROWS * RC * 2);
    __bf16* Ul  = (__bf16*)alloc((size_t)NROWS * RC * 2);
    float*  nz2 = (float*)alloc((size_t)NROWS * 4);
    float*  nm2 = (float*)alloc((size_t)NROWS * 4);

    hipMemsetAsync(nz2, 0, NROWS * sizeof(float), stream);
    hipMemsetAsync(nm2, 0, NROWS * sizeof(float), stream);

    split_bf16_kernel<<<2048, 256, 0, stream>>>(x, xh, xl, NROWS * D);
    split_bf16_kernel<<<2048, 256, 0, stream>>>(W, Wh, Wl, O * D);
    prep_At_kernel<<<(E * D * R) / 256, 256, 0, stream>>>(A, Ath, Atl);
    prep_Bst_kernel<<<(E * R * O) / 256, 256, 0, stream>>>(Bm, p, Bsh, Bsl);

    lora_u_kernel<<<NROWS / 64, 128, 0, stream>>>(xh, xl, Ath, Atl, Uh, Ul);

    dim3 gz(O / 128, NROWS / 128);
    gemm_z_kernel<<<gz, 256, 0, stream>>>(xh, xl, Wh, Wl, bias, out, nz2);

    dim3 gm(O / 64, NROWS / 128);
    lora_m_kernel<<<gm, 256, 0, stream>>>(Uh, Ul, Bsh, Bsl, nz2, nm2, out, 0);
    lora_m_kernel<<<gm, 256, 0, stream>>>(Uh, Ul, Bsh, Bsl, nz2, nm2, out, 1);
}
